// MStepModel_68753836474414
// MI455X (gfx1250) — matrix operand order repaired and checked by reading, not compiled, not run on hardware
//
#include <hip/hip_runtime.h>
#include <hip/hip_bf16.h>
#include <stdint.h>

// ---------------------------------------------------------------------------
// 2-layer GCN on MI455X (gfx1250, wave32):
//   x1     = Agg( X @ W1 ) + b1          (Agg = sym-normalized scatter-add)
//   logits = Agg( relu(x1) @ W2 ) + b2
// GEMMs: v_wmma_f32_16x16x32_bf16, f32 accumulate, f32->bf16 pair-packed
// through LDS (b128 A-tile stores and fragment reads, b32 B-tile stores). Aggregation: one wave32 per edge,
// lane-coalesced channels, global_atomic_add_f32 (h fits in 192MB L2).
// ---------------------------------------------------------------------------

typedef __attribute__((ext_vector_type(16))) __bf16    v16bf;
typedef __attribute__((ext_vector_type(2)))  __bf16    v2bf;
typedef __attribute__((ext_vector_type(8)))  float     v8f;
typedef __attribute__((ext_vector_type(4)))  uint32_t  v4u;
typedef __attribute__((ext_vector_type(4)))  float     v4f;
typedef __attribute__((ext_vector_type(2)))  float     v2f;

union FragAB { v16bf bf; v4u u[2]; };

__device__ __forceinline__ uint32_t pack_bf16(float a, float b) {
  v2f f = {a, b};
  v2bf r = __builtin_convertvector(f, v2bf);   // v_cvt_pk_bf16_f32
  return __builtin_bit_cast(uint32_t, r);
}

// ----------------------------- degree / norm -------------------------------

__global__ void k_init_deg(float* deg, int n) {
  int i = blockIdx.x * blockDim.x + threadIdx.x;
  if (i < n) deg[i] = 1.0f;  // self-loop contributes 1 to every node
}

__global__ void k_count_deg(const int* __restrict__ dst, float* deg, int e) {
  int i = blockIdx.x * blockDim.x + threadIdx.x;
  if (i < e) atomicAdd(&deg[dst[i]], 1.0f);
}

__global__ void k_rsqrt(float* deg, int n) {
  int i = blockIdx.x * blockDim.x + threadIdx.x;
  if (i < n) deg[i] = rsqrtf(deg[i]);   // in-place: deg -> deg^{-1/2}
}

// out[n][c] = bias[c]  (folds "+b" into init so aggregation is pure adds)
__global__ void k_init_bias(float* __restrict__ out, const float* __restrict__ b,
                            size_t total, int hmask) {
  size_t i = (size_t)blockIdx.x * blockDim.x + threadIdx.x;
  if (i < total) out[i] = b[i & hmask];
}

// ------------------------------ WMMA GEMM ----------------------------------
// C[M,N] = A[M,K] @ B[K,N] row-major f32; A optionally relu'd on load.
// Block: 256 thr = 8 waves. Tile BM=128, BN=64, BK=32. Wave w owns rows
// [w*16, w*16+16) x 64 cols = 4 WMMA accumulators.
// K % 32 == 0 (768, 512), N % 64 == 0 (512, 64).
// Out-of-range A rows are CLAMPED to row M-1 (not zeroed): those rows only
// feed C rows that are never stored, so no per-iteration predication needed.
// LDS tiles hold bf16 pairs packed in uint32 (K fastest): row = 16 dwords,
// padded stride 20 dwords (80B: 16B-aligned, conflict-free fragment reads).
// Epilogue: one branch per block (full tile in-range?), per-row guards only
// on the single boundary block.

#define BM 128
#define BN 64
#define BK 32
#define SA 20   // dword stride of a 32-bf16 row in sA
#define SB 20   // dword stride of a 32-bf16 column in sB (transposed)

template <bool RELU>
__global__ __launch_bounds__(256)
void k_gemm_bf16(const float* __restrict__ A, const float* __restrict__ B,
                 float* __restrict__ C, int M, int K, int N) {
  __shared__ uint32_t sA[BM * SA];   // [row][kpair]
  __shared__ uint32_t sB[BN * SB];   // [col][kpair]  (B transposed)

  const int tid  = threadIdx.x;
  const int lane = tid & 31;
  const int wave = tid >> 5;
  const int m0   = blockIdx.y * BM;
  const int n0   = blockIdx.x * BN;

  // A loader: thread -> row = tid>>1, 16-float half-row at c = (tid&1)*16.
  // Row clamped into range: no branch in the K-loop.
  const int a_r   = tid >> 1;
  const int a_c   = (tid & 1) * 16;
  const int a_gr  = m0 + a_r;
  const int a_row = (a_gr < M) ? a_gr : (M - 1);
  const float* a_src0 = A + (size_t)a_row * K + a_c;
  uint32_t* a_dst = &sA[a_r * SA + (a_c >> 1)];

  // B loader: 1024 packed units (64 cols x 16 kpairs); 4 units per thread.
  const float* b_base = B + n0;

  v8f zero = {};
  v8f acc[4];
  #pragma unroll
  for (int i = 0; i < 4; ++i) acc[i] = zero;

  // fragment addresses (packed dwords)
  const int arow = wave * 16 + (lane & 15);
  const int akb2 = (lane >= 16) ? 4 : 0;          // bf16 k-base 8 -> dword 4
  const uint32_t* ap = &sA[arow * SA + akb2];
  const int bkb2 = (lane >= 16) ? 4 : 0;          // bf16 k-base 8 -> dword 4
  const uint32_t* bp0 = &sB[(lane & 15) * SB + bkb2];

  for (int k0 = 0; k0 < K; k0 += BK) {
    // --- A tile: 4x global_load_b128 -> 8 packed dwords -> 2x ds_store_b128
    {
      const v4f* src = (const v4f*)(a_src0 + k0);
      v4f f[4];
      #pragma unroll
      for (int j = 0; j < 4; ++j) f[j] = src[j];
      if (RELU) {
        v4f z4 = {};
        #pragma unroll
        for (int j = 0; j < 4; ++j) f[j] = __builtin_elementwise_max(f[j], z4);
      }
      v4u p0, p1;
      #pragma unroll
      for (int e = 0; e < 4; ++e) {
        p0[e] = pack_bf16(f[e >> 1][(e & 1) * 2], f[e >> 1][(e & 1) * 2 + 1]);
        p1[e] = pack_bf16(f[2 + (e >> 1)][(e & 1) * 2], f[2 + (e >> 1)][(e & 1) * 2 + 1]);
      }
      *(v4u*)(a_dst)     = p0;
      *(v4u*)(a_dst + 4) = p1;
    }
    // --- B tile: per unit load 2 consecutive K of one column, pack, b32 store
    {
      #pragma unroll
      for (int i = 0; i < 4; ++i) {
        const int u   = tid + 256 * i;
        const int col = u & 63;
        const int kp  = u >> 6;
        const float* s = b_base + (size_t)(k0 + 2 * kp) * N + col;
        const float v0 = s[0];
        const float v1 = s[N];
        sB[col * SB + kp] = pack_bf16(v0, v1);
      }
    }
    __syncthreads();

    // --- fragments (ISA 7.12.2 16-bit layouts), all ds_load_b128 ---
    FragAB af;
    af.u[0] = *(const v4u*)(ap);
    af.u[1] = *(const v4u*)(ap + 8);

    #pragma unroll
    for (int n = 0; n < 4; ++n) {
      FragAB bf_;
      const uint32_t* bp = bp0 + n * 16 * SB;
      bf_.u[0] = *(const v4u*)(bp);
      bf_.u[1] = *(const v4u*)(bp + 8);
      acc[n] = __builtin_amdgcn_wmma_f32_16x16x32_bf16(
          false, af.bf, false, bf_.bf, (short)0, acc[n], false, false);
    }
    __syncthreads();
  }

  // --- epilogue: VGPR r -> M = r (+8 for lanes>=16), N = lane&15 ---
  const int rbase = wave * 16 + ((lane >= 16) ? 8 : 0);   // block-local row
  const int grow  = m0 + rbase;
  float* cbase = C + (size_t)grow * N + n0 + (lane & 15);

  if (m0 + BM <= M) {
    // fast path (all blocks except the single boundary block): no guards
    #pragma unroll
    for (int r = 0; r < 8; ++r) {
      float* cr = cbase + (size_t)r * N;
      #pragma unroll
      for (int n = 0; n < 4; ++n) cr[n * 16] = acc[n][r];
    }
  } else {
    // boundary block: one guard per row
    #pragma unroll
    for (int r = 0; r < 8; ++r) {
      if (grow + r < M) {
        float* cr = cbase + (size_t)r * N;
        #pragma unroll
        for (int n = 0; n < 4; ++n) cr[n * 16] = acc[n][r];
      }
    }
  }
}

// --------------------------- edge aggregation ------------------------------
// One wave32 per (edge or self-loop). H channels -> H/32 per lane, coalesced.
// out[dst][c] += h[src][c] * dis[src] * dis[dst]

template <int H>
__global__ __launch_bounds__(256)
void k_scatter(const int* __restrict__ srcIdx, const int* __restrict__ dstIdx,
               const float* __restrict__ dis, const float* __restrict__ h,
               float* __restrict__ out, int E, int N) {
  const int gwave = (int)(((size_t)blockIdx.x * blockDim.x + threadIdx.x) >> 5);
  const int lane  = threadIdx.x & 31;
  const int total = E + N;
  if (gwave >= total) return;

  int s, d;
  if (gwave < E) { s = srcIdx[gwave]; d = dstIdx[gwave]; }
  else           { s = d = gwave - E; }          // self-loop

  const float norm = dis[s] * dis[d];
  const float* hs = h   + (size_t)s * H;
  float*       od = out + (size_t)d * H;

  #pragma unroll
  for (int j = 0; j < H / 32; ++j) {
    const int c = lane + j * 32;                 // coalesced across the wave
    atomicAdd(&od[c], hs[c] * norm);
  }
}

// ------------------------------- launcher ----------------------------------

extern "C" void kernel_launch(void* const* d_in, const int* in_sizes, int n_in,
                              void* d_out, int out_size, void* d_ws, size_t ws_size,
                              hipStream_t stream) {
  (void)n_in; (void)out_size; (void)ws_size;

  const float* X    = (const float*)d_in[0];            // [N, 768]
  const int*   eidx = (const int*)d_in[1];              // [2, E]
  const float* W1   = (const float*)d_in[2];            // [768, 512]
  const float* b1   = (const float*)d_in[3];            // [512]
  const float* W2   = (const float*)d_in[4];            // [512, 64]
  const float* b2   = (const float*)d_in[5];            // [64]

  const int LM  = 768, HID = 512, NC = 64;
  const int N   = in_sizes[0] / LM;                     // 50000
  const int E   = in_sizes[1] / 2;                      // 800000
  const int* srcIdx = eidx;
  const int* dstIdx = eidx + E;

  // workspace layout (256B aligned)
  char* ws = (char*)d_ws;
  size_t off = 0;
  auto carve = [&](size_t bytes) { void* p = ws + off; off = (off + bytes + 255) & ~(size_t)255; return p; };
  float* dis = (float*)carve((size_t)N * sizeof(float));          // deg -> deg^-1/2
  float* h1  = (float*)carve((size_t)N * HID * sizeof(float));    // X @ W1
  float* h2  = (float*)carve((size_t)N * NC  * sizeof(float));    // relu(x1) @ W2

  float* x1     = (float*)d_out;                        // [N, 512]
  float* logits = x1 + (size_t)N * HID;                 // [N, 64]

  // 1) symmetric normalization: deg (with self-loop) -> deg^-1/2
  k_init_deg <<<(N + 255) / 256, 256, 0, stream>>>(dis, N);
  k_count_deg<<<(E + 255) / 256, 256, 0, stream>>>(dstIdx, dis, E);
  k_rsqrt    <<<(N + 255) / 256, 256, 0, stream>>>(dis, N);

  // 2) h1 = X @ W1   (bf16 WMMA, f32 accumulate)
  {
    dim3 grid(HID / BN, (N + BM - 1) / BM);
    k_gemm_bf16<false><<<grid, 256, 0, stream>>>(X, W1, h1, N, LM, HID);
  }

  // 3) x1 = b1; x1[dst] += h1[src] * norm   (edges + self-loops)
  {
    size_t tot = (size_t)N * HID;
    k_init_bias<<<(unsigned)((tot + 255) / 256), 256, 0, stream>>>(x1, b1, tot, HID - 1);
    size_t waves = (size_t)E + N;
    k_scatter<512><<<(unsigned)((waves * 32 + 255) / 256), 256, 0, stream>>>(
        srcIdx, dstIdx, dis, h1, x1, E, N);
  }

  // 4) h2 = relu(x1) @ W2   (ReLU fused into the bf16 tile load)
  {
    dim3 grid(NC / BN, (N + BM - 1) / BM);
    k_gemm_bf16<true><<<grid, 256, 0, stream>>>(x1, W2, h2, N, HID, NC);
  }

  // 5) logits = b2; logits[dst] += h2[src] * norm
  {
    size_t tot = (size_t)N * NC;
    k_init_bias<<<(unsigned)((tot + 255) / 256), 256, 0, stream>>>(logits, b2, tot, NC - 1);
    size_t waves = (size_t)E + N;
    k_scatter<64><<<(unsigned)((waves * 32 + 255) / 256), 256, 0, stream>>>(
        srcIdx, dstIdx, dis, h2, logits, E, N);
  }
}
